// GGNNClassifierFeatsNoEmb_4337916969199
// MI455X (gfx1250) — matrix operand order repaired and checked by reading, not compiled, not run on hardware
//
#include <hip/hip_runtime.h>
#include <hip/hip_bf16.h>

typedef __attribute__((ext_vector_type(16))) __bf16 v16bf;
typedef __attribute__((ext_vector_type(4)))  __bf16 v4bf;
typedef __attribute__((ext_vector_type(8)))  float  v8f;

#define IN_DIM   128
#define GATE_DIM 384
#define NTYPES   3
#define NGRAPHS  64
#define PROJ_DIM 128

// ---------------- WMMA fragment helpers (bf16 16x16x32, wave32) ----------------
// A (16x32, row-major source, lane = m in 0..15 duplicated over half-wave):
//   element e : K = hi*16 + e
__device__ __forceinline__ v16bf load_A_f32(const float* base, int ld, int kc, int lane) {
  int m  = lane & 15;
  int hi = lane >> 4;
  const float* p = base + (size_t)m * ld + kc * 32;
  v16bf a;
#pragma unroll
  for (int e = 0; e < 8; ++e) {
    a[e]     = (__bf16)p[hi * 16 + e];
    a[8 + e] = (__bf16)p[hi * 16 + 8 + e];
  }
  return a;
}

// B (32x16 from row-major weight W[o][i], lane holds column n = lane&15):
//   element e : K = hi*16 + e  -> 16 contiguous bf16 (32B) per lane
__device__ __forceinline__ v16bf load_B(const __bf16* W, int colTile, int kc, int lane) {
  int n  = lane & 15;
  int hi = lane >> 4;
  const __bf16* p = W + ((size_t)(colTile * 16 + n)) * IN_DIM + kc * 32 + hi * 16;
  return *(const v16bf*)p;
}

__device__ __forceinline__ v8f wmma_bf16(v16bf a, v16bf b, v8f c) {
  return __builtin_amdgcn_wmma_f32_16x16x32_bf16(false, a, false, b, (short)0, c, false, false);
}

__device__ __forceinline__ void atomAddF(float* p, float v) {
  unsafeAtomicAdd(p, v);  // global_atomic_add_f32
}

// ---------------- Kernels ----------------

// x[n] = concat(type_emb[x_type[n]], tok_emb[x_tok[n]], x_small[n])
__global__ void k_build_x(const int* __restrict__ xtype, const int* __restrict__ xtok,
                          const float* __restrict__ xsmall,
                          const float* __restrict__ type_emb, const float* __restrict__ tok_emb,
                          float* __restrict__ x, int N) {
  int idx = blockIdx.x * blockDim.x + threadIdx.x;
  if (idx >= N * IN_DIM) return;
  int n = idx >> 7, c = idx & 127;
  float v;
  if (c < 32)       v = type_emb[(size_t)xtype[n] * 32 + c];
  else if (c < 64)  v = tok_emb[(size_t)xtok[n] * 32 + (c - 32)];
  else              v = xsmall[(size_t)n * 64 + (c - 64)];
  x[idx] = v;
}

__global__ void k_f32_to_bf16(const float* __restrict__ in, __bf16* __restrict__ out, int n) {
  int i = blockIdx.x * blockDim.x + threadIdx.x;
  if (i < n) out[i] = (__bf16)in[i];
}

__global__ void k_zero_f32(float* __restrict__ p, int n) {
  int i = blockIdx.x * blockDim.x + threadIdx.x;
  if (i < n) p[i] = 0.0f;
}

// h_all[t][n][o] = sum_i x[n][i] * msg_W[t][o][i] + msg_b[t][o], stored bf16.
// One workgroup (8 waves) per 16-node tile; wave w computes output tile w (of 8)
// for each of the 3 edge types. 12 WMMA per wave.
__global__ void k_msg_transform(const float* __restrict__ x, const __bf16* __restrict__ Wb,
                                const float* __restrict__ msg_b, __bf16* __restrict__ h_all, int N) {
  int tile = blockIdx.x;
  int lane = threadIdx.x & 31;
  int wave = threadIdx.x >> 5;  // 0..7 -> output col tile
  const float* xrow = x + (size_t)tile * 16 * IN_DIM;

  v16bf a[4];
#pragma unroll
  for (int kc = 0; kc < 4; ++kc) a[kc] = load_A_f32(xrow, IN_DIM, kc, lane);

  int colN = lane & 15;
  int hi   = lane >> 4;
#pragma unroll
  for (int t = 0; t < NTYPES; ++t) {
    const __bf16* W = Wb + (size_t)t * IN_DIM * IN_DIM;
    v8f acc = {};
#pragma unroll
    for (int kc = 0; kc < 4; ++kc) acc = wmma_bf16(a[kc], load_B(W, wave, kc, lane), acc);
    int col = wave * 16 + colN;
    float bias = msg_b[t * IN_DIM + col];
    __bf16* out = h_all + ((size_t)t * N + (size_t)tile * 16) * IN_DIM + col;
#pragma unroll
    for (int r = 0; r < 8; ++r) {
      int m = r + 8 * hi;
      out[(size_t)m * IN_DIM] = (__bf16)(acc[r] + bias);
    }
  }
}

// agg[dst[e]] += h_all[edge_type[e]][src[e]]; thread handles 4 columns of one edge.
__global__ void k_edge_scatter(const __bf16* __restrict__ h_all,
                               const int* __restrict__ src, const int* __restrict__ dst,
                               const int* __restrict__ etype,
                               float* __restrict__ agg, int E, int N) {
  int idx = blockIdx.x * blockDim.x + threadIdx.x;
  int e  = idx >> 5;
  if (e >= E) return;
  int c4 = (idx & 31) * 4;
  int t = etype[e], s = src[e], d = dst[e];
  v4bf mv = *(const v4bf*)(h_all + ((size_t)t * N + s) * IN_DIM + c4);
  float* ap = agg + (size_t)d * IN_DIM + c4;
#pragma unroll
  for (int j = 0; j < 4; ++j) atomAddF(ap + j, (float)mv[j]);
}

// Fused GRU step for one 16-node tile. Waves 0-3: gi = agg @ W_ih^T (24 col tiles);
// waves 4-7: gh = x @ W_hh^T. Results in LDS, then elementwise gates, x updated in place.
__global__ void k_gru_step(const float* __restrict__ agg, float* __restrict__ x,
                           const __bf16* __restrict__ Wihb, const __bf16* __restrict__ Whhb,
                           const float* __restrict__ b_ih, const float* __restrict__ b_hh, int N) {
  __shared__ float sbuf[2 * 16 * GATE_DIM];  // 48 KB
  float* sgi = sbuf;
  float* sgh = sbuf + 16 * GATE_DIM;

  int tile = blockIdx.x;
  int lane = threadIdx.x & 31;
  int wave = threadIdx.x >> 5;
  bool isGh = wave >= 4;
  int  w4   = wave & 3;

  const float*  Arow = (isGh ? x : agg) + (size_t)tile * 16 * IN_DIM;
  const __bf16* W    = isGh ? Whhb : Wihb;
  const float*  bias = isGh ? b_hh : b_ih;
  float*        out  = isGh ? sgh : sgi;

  v16bf a[4];
#pragma unroll
  for (int kc = 0; kc < 4; ++kc) a[kc] = load_A_f32(Arow, IN_DIM, kc, lane);

  int colN = lane & 15;
  int hi   = lane >> 4;
#pragma unroll
  for (int rep = 0; rep < 6; ++rep) {
    int ct = w4 * 6 + rep;  // 0..23 output col tiles of 384
    v8f acc = {};
#pragma unroll
    for (int kc = 0; kc < 4; ++kc) acc = wmma_bf16(a[kc], load_B(W, ct, kc, lane), acc);
    int col = ct * 16 + colN;
    float bi = bias[col];
#pragma unroll
    for (int r = 0; r < 8; ++r) {
      int m = r + 8 * hi;
      out[m * GATE_DIM + col] = acc[r] + bi;
    }
  }
  __syncthreads();

  // 16 nodes x 128 cols = 2048 elements; 256 threads x 8
#pragma unroll
  for (int it = 0; it < 8; ++it) {
    int idx = threadIdx.x + it * 256;
    int m = idx >> 7, c = idx & 127;
    float gir = sgi[m * GATE_DIM + c];
    float giz = sgi[m * GATE_DIM + 128 + c];
    float gin = sgi[m * GATE_DIM + 256 + c];
    float ghr = sgh[m * GATE_DIM + c];
    float ghz = sgh[m * GATE_DIM + 128 + c];
    float ghn = sgh[m * GATE_DIM + 256 + c];
    float r = 1.0f / (1.0f + __expf(-(gir + ghr)));
    float z = 1.0f / (1.0f + __expf(-(giz + ghz)));
    float nn = tanhf(gin + r * ghn);
    size_t g = ((size_t)tile * 16 + m) * IN_DIM + c;
    float h = x[g];
    x[g] = (1.0f - z) * nn + z * h;
  }
}

// segment sums / counts per graph
__global__ void k_pool_scatter(const float* __restrict__ x, const int* __restrict__ batch,
                               float* __restrict__ sums, float* __restrict__ counts, int N) {
  int idx = blockIdx.x * blockDim.x + threadIdx.x;
  int n = idx >> 5;
  if (n >= N) return;
  int c4 = (idx & 31) * 4;
  int g = batch[n];
  const float* xp = x + (size_t)n * IN_DIM + c4;
  float* sp = sums + (size_t)g * IN_DIM + c4;
#pragma unroll
  for (int j = 0; j < 4; ++j) atomAddF(sp + j, xp[j]);
  if ((idx & 31) == 0) atomAddF(&counts[g], 1.0f);
}

// h = relu(pooled @ W1^T + b1)
__global__ void k_proj1(const float* __restrict__ sums, const float* __restrict__ counts,
                        const float* __restrict__ W1, const float* __restrict__ b1,
                        float* __restrict__ hproj) {
  int idx = blockIdx.x * blockDim.x + threadIdx.x;
  if (idx >= NGRAPHS * PROJ_DIM) return;
  int g = idx >> 7, p = idx & 127;
  float inv = 1.0f / fmaxf(counts[g], 1.0f);
  float acc = b1[p];
  const float* sr = sums + (size_t)g * IN_DIM;
  const float* wr = W1 + (size_t)p * IN_DIM;
#pragma unroll 4
  for (int i = 0; i < IN_DIM; ++i) acc += sr[i] * inv * wr[i];
  hproj[idx] = fmaxf(acc, 0.0f);
}

__global__ void k_proj2(const float* __restrict__ hproj, const float* __restrict__ W2,
                        const float* __restrict__ b2, float* __restrict__ out) {
  int g = threadIdx.x;
  if (g >= NGRAPHS) return;
  float acc = b2[0];
  const float* hr = hproj + (size_t)g * PROJ_DIM;
#pragma unroll 4
  for (int p = 0; p < PROJ_DIM; ++p) acc += hr[p] * W2[p];
  out[g] = acc;
}

// ---------------- Host launcher ----------------
extern "C" void kernel_launch(void* const* d_in, const int* in_sizes, int n_in,
                              void* d_out, int out_size, void* d_ws, size_t ws_size,
                              hipStream_t stream) {
  const int*   xtype   = (const int*)  d_in[0];
  const int*   xtok    = (const int*)  d_in[1];
  const float* xsmall  = (const float*)d_in[2];
  const int*   eidx    = (const int*)  d_in[3];
  const int*   etype   = (const int*)  d_in[4];
  const int*   batch   = (const int*)  d_in[5];
  const float* type_emb= (const float*)d_in[6];
  const float* tok_emb = (const float*)d_in[7];
  const float* msg_W   = (const float*)d_in[8];
  const float* msg_b   = (const float*)d_in[9];
  const float* W_ih    = (const float*)d_in[10];
  const float* W_hh    = (const float*)d_in[11];
  const float* b_ih    = (const float*)d_in[12];
  const float* b_hh    = (const float*)d_in[13];
  const float* proj_W1 = (const float*)d_in[14];
  const float* proj_b1 = (const float*)d_in[15];
  const float* proj_W2 = (const float*)d_in[16];
  const float* proj_b2 = (const float*)d_in[17];

  const int N = in_sizes[0];
  const int E = in_sizes[4];
  const int* src = eidx;
  const int* dst = eidx + E;
  const int NT = (N + 15) / 16;

  // workspace carve-up (256B aligned regions)
  char* ws = (char*)d_ws;
  size_t off = 0;
  auto carve = [&](size_t bytes) -> char* {
    char* p = ws + off;
    off = (off + bytes + 255) & ~(size_t)255;
    return p;
  };
  float*  x      = (float*) carve((size_t)(N + 16) * IN_DIM * sizeof(float));
  float*  agg    = (float*) carve((size_t)(N + 16) * IN_DIM * sizeof(float));
  __bf16* h_all  = (__bf16*)carve((size_t)NTYPES * (N + 16) * IN_DIM * sizeof(__bf16));
  __bf16* Wmsgb  = (__bf16*)carve((size_t)NTYPES * IN_DIM * IN_DIM * sizeof(__bf16));
  __bf16* Wihb   = (__bf16*)carve((size_t)GATE_DIM * IN_DIM * sizeof(__bf16));
  __bf16* Whhb   = (__bf16*)carve((size_t)GATE_DIM * IN_DIM * sizeof(__bf16));
  float*  sums   = (float*) carve((size_t)NGRAPHS * IN_DIM * sizeof(float));
  float*  counts = (float*) carve((size_t)NGRAPHS * sizeof(float));
  float*  hproj  = (float*) carve((size_t)NGRAPHS * PROJ_DIM * sizeof(float));

  const int TPB = 256;
  int nx = N * IN_DIM;

  k_build_x<<<(nx + TPB - 1) / TPB, TPB, 0, stream>>>(xtype, xtok, xsmall, type_emb, tok_emb, x, N);
  k_f32_to_bf16<<<(NTYPES * IN_DIM * IN_DIM + TPB - 1) / TPB, TPB, 0, stream>>>(msg_W, Wmsgb, NTYPES * IN_DIM * IN_DIM);
  k_f32_to_bf16<<<(GATE_DIM * IN_DIM + TPB - 1) / TPB, TPB, 0, stream>>>(W_ih, Wihb, GATE_DIM * IN_DIM);
  k_f32_to_bf16<<<(GATE_DIM * IN_DIM + TPB - 1) / TPB, TPB, 0, stream>>>(W_hh, Whhb, GATE_DIM * IN_DIM);

  for (int step = 0; step < 8; ++step) {
    k_msg_transform<<<NT, TPB, 0, stream>>>(x, Wmsgb, msg_b, h_all, N);
    k_zero_f32<<<(nx + TPB - 1) / TPB, TPB, 0, stream>>>(agg, nx);
    k_edge_scatter<<<((size_t)E * 32 + TPB - 1) / TPB, TPB, 0, stream>>>(h_all, src, dst, etype, agg, E, N);
    k_gru_step<<<NT, TPB, 0, stream>>>(agg, x, Wihb, Whhb, b_ih, b_hh, N);
  }

  k_zero_f32<<<(NGRAPHS * IN_DIM + TPB - 1) / TPB, TPB, 0, stream>>>(sums, NGRAPHS * IN_DIM);
  k_zero_f32<<<1, TPB, 0, stream>>>(counts, NGRAPHS);
  k_pool_scatter<<<((size_t)N * 32 + TPB - 1) / TPB, TPB, 0, stream>>>(x, batch, sums, counts, N);
  k_proj1<<<(NGRAPHS * PROJ_DIM + TPB - 1) / TPB, TPB, 0, stream>>>(sums, counts, proj_W1, proj_b1, hproj);
  k_proj2<<<1, 64, 0, stream>>>(hproj, proj_W2, proj_b2, (float*)d_out);
}
